// EdgeWeightedQBAF_38869454029395
// MI455X (gfx1250) — matrix operand order repaired and checked by reading, not compiled, not run on hardware
//
#include <hip/hip_runtime.h>
#include <hip/hip_bf16.h>

// Problem constants (match reference)
#define BATCH 4096
#define NF    2048   // input features
#define NN    1024   // hidden neurons
#define NT    8      // targets
#define NNZ1  65536
#define NNZ2  8192

typedef __attribute__((ext_vector_type(16))) __bf16          v16bf;
typedef __attribute__((ext_vector_type(16))) unsigned short  v16u;
typedef __attribute__((ext_vector_type(8)))  float           v8f;

__device__ __forceinline__ unsigned short f2bf_bits(float f) {
    // round-to-nearest-even fp32 -> bf16 (bit pattern)
    unsigned u = __float_as_uint(f);
    u += 0x7FFFu + ((u >> 16) & 1u);
    return (unsigned short)(u >> 16);
}

// gfx1250 async memory->LDS staging (ASYNCcnt-tracked, no VGPR round trip)
#define ASYNC_LDS_B128(ldsoff, gaddr)                                         \
    asm volatile("global_load_async_to_lds_b128 %0, %1, off"                  \
                 :: "v"(ldsoff), "v"(gaddr) : "memory")

__device__ __forceinline__ unsigned lds_off32(const void* p) {
    // generic pointer to LDS: low 32 bits are the LDS byte offset
    return (unsigned)(unsigned long long)p;
}

// ---------------------------------------------------------------- utilities
__global__ void zero_f32_kernel(float* __restrict__ p, int n) {
    int i = blockIdx.x * blockDim.x + threadIdx.x;
    if (i < n) p[i] = 0.0f;
}

__global__ void scatter_add_kernel(const float* __restrict__ w,
                                   const int* __restrict__ rows,
                                   const int* __restrict__ cols,
                                   float* __restrict__ dense,
                                   int nnz, int ldk) {
    int e = blockIdx.x * blockDim.x + threadIdx.x;
    if (e < nnz) atomicAdd(&dense[rows[e] * ldk + cols[e]], w[e]);
}

__global__ void cvt_bf16_kernel(const float* __restrict__ src,
                                unsigned short* __restrict__ dst, int n) {
    int i = blockIdx.x * blockDim.x + threadIdx.x;
    if (i < n) dst[i] = f2bf_bits(src[i]);
}

// ---------------------------------------------------------------- GEMM1 (WMMA)
// H[4096x1024] = sigmoid(Xh[4096x2048] * W1h[1024x2048]^T + b1)
// Block: 256 threads = 8 waves, 4(M) x 2(N) wave grid -> 64(M) x 64(N) C tile.
// Each wave computes a 16x32 slice: two 16x16 accumulators sharing one A frag.
// K staged in steps of 32 through a double-buffered LDS tile filled by
// global_load_async_to_lds_b128, pipelined one tile ahead (s_wait_asynccnt 2).
__launch_bounds__(256, 2)
__global__ void gemm1_wmma_kernel(const unsigned short* __restrict__ Xh,
                                  const unsigned short* __restrict__ W1h,
                                  const float* __restrict__ b1,
                                  float* __restrict__ H) {
    __shared__ __align__(16) unsigned short lsA[2][64 * 32]; // [m][k] per buffer
    __shared__ __align__(16) unsigned short lsB[2][64 * 32]; // [n][k] per buffer

    const int lane = threadIdx.x & 31;
    const int wave = threadIdx.x >> 5;
    const int wm   = wave >> 1;            // 0..3 -> M sub-tile (16 rows)
    const int wn   = wave & 1;             // 0..1 -> N half (32 cols = 2 tiles)
    const int nBase = blockIdx.x * 64;
    const int mBase = blockIdx.y * 64;

    // cooperative staging: 16B/thread for each of A and B (4 KB tiles)
    const int tRow = threadIdx.x >> 2;            // 0..63
    const int tCol = (threadIdx.x & 3) * 8;       // 16B chunks

    const unsigned short* gA = &Xh[(size_t)(mBase + tRow) * NF + tCol];
    const unsigned short* gB = &W1h[(size_t)(nBase + tRow) * NF + tCol];
    const unsigned ldsA[2] = { lds_off32(&lsA[0][tRow * 32 + tCol]),
                               lds_off32(&lsA[1][tRow * 32 + tCol]) };
    const unsigned ldsB[2] = { lds_off32(&lsB[0][tRow * 32 + tCol]),
                               lds_off32(&lsB[1][tRow * 32 + tCol]) };

    // fragment coordinates (16-bit WMMA lanes; contiguous K per lane half, same for A and B)
    const int am = wm * 16 + (lane & 15);         // A row (M)
    const int kh = (lane >> 4) * 16;              // A: K=0..15 or 16..31
    const int bn = wn * 32 + (lane & 15);         // B col (N), first n-tile
    const int bk = (lane >> 4) * 16;              // B: K=0..15 or 16..31

    v8f acc0 = {};
    v8f acc1 = {};

    // prologue: stage tile 0 into buffer 0
    ASYNC_LDS_B128(ldsA[0], (unsigned long long)gA);
    ASYNC_LDS_B128(ldsB[0], (unsigned long long)gB);

    for (int k0 = 0, it = 0; k0 < NF; k0 += 32, ++it) {
        const int buf = it & 1;
        if (k0 + 32 < NF) {
            // stage next tile into the other buffer (drained by last barrier)
            ASYNC_LDS_B128(ldsA[buf ^ 1], (unsigned long long)(gA + k0 + 32));
            ASYNC_LDS_B128(ldsB[buf ^ 1], (unsigned long long)(gB + k0 + 32));
            // async ops complete in order: <=2 outstanding => current tile landed
            asm volatile("s_wait_asynccnt 0x2" ::: "memory");
        } else {
            asm volatile("s_wait_asynccnt 0x0" ::: "memory");
        }
        __syncthreads();

        // A fragment (shared by both WMMAs):
        // lanes 0-15 hold K 0..15, lanes 16-31 hold 16..31
        v16u au, bu0, bu1;
        const unsigned short* pa = &lsA[buf][am * 32 + kh];
        #pragma unroll
        for (int j = 0; j < 8; ++j) au[j] = pa[j];
        #pragma unroll
        for (int j = 0; j < 8; ++j) au[8 + j] = pa[8 + j];

        // B fragments: lane holds 16 consecutive K of column bn / bn+16
        const unsigned short* pb0 = &lsB[buf][bn * 32 + bk];
        const unsigned short* pb1 = pb0 + 16 * 32;
        #pragma unroll
        for (int j = 0; j < 16; ++j) bu0[j] = pb0[j];
        #pragma unroll
        for (int j = 0; j < 16; ++j) bu1[j] = pb1[j];

        v16bf a  = __builtin_bit_cast(v16bf, au);
        v16bf b0 = __builtin_bit_cast(v16bf, bu0);
        v16bf b1 = __builtin_bit_cast(v16bf, bu1);
        acc0 = __builtin_amdgcn_wmma_f32_16x16x32_bf16(
            false, a, false, b0, (short)0, acc0, false, false);
        acc1 = __builtin_amdgcn_wmma_f32_16x16x32_bf16(
            false, a, false, b1, (short)0, acc1, false, false);
        __syncthreads();
    }

    // epilogue: C layout -> VGPR r: lanes 0-15 M=r, lanes 16-31 M=8+r; N = lane&15
    const int n0   = nBase + wn * 32 + (lane & 15);
    const int mRow = mBase + wm * 16 + ((lane >> 4) * 8);
    const float bias0 = b1[n0];
    const float bias1 = b1[n0 + 16];
    #pragma unroll
    for (int r = 0; r < 8; ++r) {
        float v0 = acc0[r] + bias0;
        float v1 = acc1[r] + bias1;
        v0 = __builtin_amdgcn_rcpf(1.0f + __expf(-v0));  // fast sigmoid
        v1 = __builtin_amdgcn_rcpf(1.0f + __expf(-v1));
        H[(size_t)(mRow + r) * NN + n0]      = v0;
        H[(size_t)(mRow + r) * NN + n0 + 16] = v1;
    }
}

// ---------------------------------------------------------------- GEMM2 (narrow N=8)
__global__ void gemm2_kernel(const float* __restrict__ H,
                             const float* __restrict__ W2,
                             const float* __restrict__ b2,
                             float* __restrict__ out) {
    int idx = blockIdx.x * blockDim.x + threadIdx.x;   // BATCH*NT
    if (idx >= BATCH * NT) return;
    int b = idx >> 3;
    int o = idx & 7;
    const float4* hr = reinterpret_cast<const float4*>(&H[(size_t)b * NN]);
    const float4* wr = reinterpret_cast<const float4*>(&W2[(size_t)o * NN]);
    float s = 0.0f;
    #pragma unroll 4
    for (int k = 0; k < NN / 4; ++k) {
        float4 h4 = hr[k];
        float4 w4 = wr[k];
        s += h4.x * w4.x + h4.y * w4.y + h4.z * w4.z + h4.w * w4.w;
    }
    out[idx] = s + b2[o];
}

// ---------------------------------------------------------------- launch
extern "C" void kernel_launch(void* const* d_in, const int* in_sizes, int n_in,
                              void* d_out, int out_size, void* d_ws, size_t ws_size,
                              hipStream_t stream) {
    // inputs in setup_inputs() order
    const float* x         = (const float*)d_in[0];   // [4096 x 2048]
    const float* w1        = (const float*)d_in[1];   // [65536]
    const float* b1        = (const float*)d_in[2];   // [1024]
    const float* w2        = (const float*)d_in[3];   // [8192]
    const float* b2        = (const float*)d_in[4];   // [8]
    const int*   conn1_out = (const int*)d_in[5];
    const int*   conn1_in  = (const int*)d_in[6];
    const int*   conn2_out = (const int*)d_in[7];
    const int*   conn2_in  = (const int*)d_in[8];
    float* out = (float*)d_out;                       // [4096 x 8]

    // workspace layout (all power-of-two sized, 16B aligned)
    char* ws = (char*)d_ws;
    float*          W1f = (float*)(ws);                                   // 8 MB
    float*          W2f = (float*)(ws + (size_t)NN * NF * 4);             // 32 KB
    unsigned short* W1h = (unsigned short*)(ws + (size_t)NN * NF * 4 + (size_t)NT * NN * 4); // 4 MB
    unsigned short* Xh  = (unsigned short*)((char*)W1h + (size_t)NN * NF * 2);               // 16 MB
    float*          Hbuf= (float*)((char*)Xh + (size_t)BATCH * NF * 2);                      // 16 MB

    const int T = 256;

    // 1) zero dense weight buffers
    zero_f32_kernel<<<(NN * NF + T - 1) / T, T, 0, stream>>>(W1f, NN * NF);
    zero_f32_kernel<<<(NT * NN + T - 1) / T, T, 0, stream>>>(W2f, NT * NN);

    // 2) scatter sparse weights into dense (atomicAdd handles duplicate edges)
    scatter_add_kernel<<<(NNZ1 + T - 1) / T, T, 0, stream>>>(w1, conn1_out, conn1_in, W1f, NNZ1, NF);
    scatter_add_kernel<<<(NNZ2 + T - 1) / T, T, 0, stream>>>(w2, conn2_out, conn2_in, W2f, NNZ2, NN);

    // 3) convert W1 dense and x to bf16
    cvt_bf16_kernel<<<(NN * NF + T - 1) / T, T, 0, stream>>>(W1f, W1h, NN * NF);
    cvt_bf16_kernel<<<(BATCH * NF + T - 1) / T, T, 0, stream>>>(x, Xh, BATCH * NF);

    // 4) H = sigmoid(Xh * W1h^T + b1) via WMMA bf16, async-LDS pipelined
    dim3 g1(NN / 64, BATCH / 64);
    gemm1_wmma_kernel<<<g1, T, 0, stream>>>(Xh, W1h, b1, Hbuf);

    // 5) out = H * W2^T + b2 (N=8, fp32 dot products)
    gemm2_kernel<<<(BATCH * NT + T - 1) / T, T, 0, stream>>>(Hbuf, W2f, b2, out);
}
